// Head_1348619731556
// MI455X (gfx1250) — matrix operand order repaired and checked by reading, not compiled, not run on hardware
//
#include <hip/hip_runtime.h>
#include <hip/hip_bf16.h>

// ---------------------------------------------------------------------------
// Problem dims (match setup_inputs): B=8, T=4096, C=1024, H=128
// ---------------------------------------------------------------------------
#define DIM_B 8
#define DIM_T 4096
#define DIM_C 1024
#define DIM_H 128
#define ROWS  (DIM_B * DIM_T)   // 32768

typedef __attribute__((ext_vector_type(16))) __bf16 v16bf;
typedef __attribute__((ext_vector_type(8)))  float  v8f;
typedef __attribute__((ext_vector_type(4)))  unsigned int u32x4;
typedef __attribute__((ext_vector_type(8)))  int          i32x8;
typedef __attribute__((ext_vector_type(4)))  int          i32x4;

struct __align__(16) U4 { unsigned v[4]; };
union Frag { U4 u[2]; v16bf v; };   // 32 bytes = one 16-element bf16 WMMA operand

__device__ __forceinline__ unsigned short f2bf(float f) {
    unsigned u = __float_as_uint(f);
    unsigned r = u + 0x7FFFu + ((u >> 16) & 1u);   // round-to-nearest-even
    return (unsigned short)(r >> 16);
}

// ---------------------------------------------------------------------------
// Kernel 1: projection GEMM  out[M=32768, N=128] = X[M,1024] * W^T (W is [128,1024])
// Block tile 64(M) x 128(N), K-step 32. 8 waves; wave computes 16x64.
// ---------------------------------------------------------------------------
template <int WRITE_BF>
__global__ __launch_bounds__(256) void proj_gemm(
    const float* __restrict__ X, const float* __restrict__ W, void* __restrict__ outP)
{
    __shared__ __align__(16) unsigned short xs[64 * 32];    // A tile (bf16, row-major)
    __shared__ __align__(16) unsigned short wsm[128 * 32];  // W rows = B^T tile

    const int tid  = threadIdx.x;
    const int wave = tid >> 5, lane = tid & 31;
    const int g = lane >> 4, ln = lane & 15;
    const int row0  = blockIdx.x * 64;
    const int mSub  = wave & 3;   // which 16-row slab
    const int nHalf = wave >> 2;  // which 64-col half

    v8f acc[4] = {};

    for (int k0 = 0; k0 < DIM_C; k0 += 32) {
        __syncthreads();
        { // stage X 64x32 (f32 -> bf16)
            int r  = tid >> 2;
            int kk = (tid & 3) * 8;
            const float* src = X + (size_t)(row0 + r) * DIM_C + k0 + kk;
            unsigned short* dst = &xs[r * 32 + kk];
            #pragma unroll
            for (int e = 0; e < 8; ++e) dst[e] = f2bf(src[e]);
        }
        { // stage W 128x32 (f32 -> bf16)
            int r  = tid >> 1;
            int kk = (tid & 1) * 16;
            const float* src = W + (size_t)r * DIM_C + k0 + kk;
            unsigned short* dst = &wsm[r * 32 + kk];
            #pragma unroll
            for (int e = 0; e < 16; ++e) dst[e] = f2bf(src[e]);
        }
        __syncthreads();

        Frag a; // A 16x32: lane=M, elem e -> K=16g+e
        {
            const unsigned short* base = &xs[(mSub * 16 + ln) * 32];
            a.u[0] = *(const U4*)(base + 16 * g);
            a.u[1] = *(const U4*)(base + 16 * g + 8);
        }
        #pragma unroll
        for (int t = 0; t < 4; ++t) { // B 32x16: lane=N, elem e -> K=16g+e
            Frag bf;
            const unsigned short* bb = &wsm[(nHalf * 64 + t * 16 + ln) * 32 + 16 * g];
            bf.u[0] = *(const U4*)(bb);
            bf.u[1] = *(const U4*)(bb + 8);
            acc[t] = __builtin_amdgcn_wmma_f32_16x16x32_bf16(
                false, a.v, false, bf.v, (short)0, acc[t], false, false);
        }
    }

    // C/D layout: elem j -> M = j + 8g, N = lane%16
    #pragma unroll
    for (int t = 0; t < 4; ++t) {
        const int col = nHalf * 64 + t * 16 + ln;
        #pragma unroll
        for (int j = 0; j < 8; ++j) {
            const size_t off = (size_t)(row0 + mSub * 16 + j + 8 * g) * DIM_H + col;
            float r = acc[t][j];
            if (WRITE_BF) ((unsigned short*)outP)[off] = f2bf(r);
            else          ((float*)outP)[off] = r;
        }
    }
}

// ---------------------------------------------------------------------------
// Kernel 2: RoPE on q,k (f32 in) -> bf16 out. One thread per (row, pair).
// ---------------------------------------------------------------------------
__global__ __launch_bounds__(256) void rope_cvt(
    const float* __restrict__ qf, const float* __restrict__ kf,
    unsigned short* __restrict__ qb, unsigned short* __restrict__ kb, int total)
{
    int idx = blockIdx.x * 256 + threadIdx.x;
    if (idx >= total) return;
    int row = idx >> 6;        // / (H/2)
    int p   = idx & 63;
    int t   = row & (DIM_T - 1);
    // freq = THETA^(-(2p)/128) ; ln(10000) = 9.210340371976184
    float inv = __expf(-9.2103403719761841f * (float)(2 * p) * (1.0f / 128.0f));
    float ang = (float)t * inv;
    float s, c;
    sincosf(ang, &s, &c);
    size_t off = (size_t)row * DIM_H + 2 * p;
    float a = qf[off], b = qf[off + 1];
    qb[off]     = f2bf(a * c - b * s);
    qb[off + 1] = f2bf(a * s + b * c);
    a = kf[off]; b = kf[off + 1];
    kb[off]     = f2bf(a * c - b * s);
    kb[off + 1] = f2bf(a * s + b * c);
}

// ---------------------------------------------------------------------------
// Kernel 3: causal flash attention. Grid = B * (T/128); 8 waves per block.
// Wave w owns Q rows [q0, q0+16). Block stages 32 keys of K (via TDM) and
// V (manual transpose) per iteration.
// ---------------------------------------------------------------------------
__global__ __launch_bounds__(256) void flash_attn(
    const unsigned short* __restrict__ qb,
    const unsigned short* __restrict__ kb,
    const unsigned short* __restrict__ vb,
    float* __restrict__ out)
{
    __shared__ __align__(16) unsigned short kS[32 * 128];      // K block, row-major [key][h]
    __shared__ __align__(16) unsigned short vT[128 * 32];      // V block transposed [h][key]
    __shared__ __align__(16) unsigned short pS[8 * 16 * 32];   // per-wave P scratch [m][k]

    const int tid  = threadIdx.x;
    const int wave = tid >> 5, lane = tid & 31;
    const int g = lane >> 4, ln = lane & 15;
    const int b    = blockIdx.x >> 5;   // / (T/128)
    const int qblk = blockIdx.x & 31;
    const int q0   = qblk * 128 + wave * 16;
    const size_t rowBase = (size_t)b * DIM_T;

    // Q fragments: 16 rows x 128 h = 4 K-chunks of 32
    Frag qfr[4];
    {
        const unsigned short* qrow = qb + (rowBase + q0 + ln) * DIM_H;
        #pragma unroll
        for (int hc = 0; hc < 4; ++hc) {
            qfr[hc].u[0] = *(const U4*)(qrow + hc * 32 + 16 * g);
            qfr[hc].u[1] = *(const U4*)(qrow + hc * 32 + 16 * g + 8);
        }
    }

    v8f oacc[8] = {};
    float lsum[8], mrow[8];
    #pragma unroll
    for (int j = 0; j < 8; ++j) { lsum[j] = 0.0f; mrow[j] = -3.0e38f; }
    const float scale = 0.08838834764831845f;   // 1/sqrt(128)

    const int nIter = (qblk + 1) * 4;           // keys up to block's last q row
    for (int it = 0; it < nIter; ++it) {
        const int kv0 = it * 32;
        __syncthreads();

        if (wave == 0) {
            // Tensor Data Mover: 2D tile, 32 rows (keys) x 128 elems (h), 2B/elem,
            // row stride 128 elems, into kS. D# per ISA cdna5 ch.8.
            const unsigned short* gsrc = kb + (rowBase + kv0) * DIM_H;
            unsigned long long ga = (unsigned long long)gsrc;
            unsigned ldsOff = (unsigned)(size_t)&kS[0];   // low 32 bits of LDS aperture addr
            u32x4 g0;
            g0[0] = 1u;                                   // count=1, user mode
            g0[1] = ldsOff;                               // lds_addr
            g0[2] = (unsigned)ga;                         // global_addr[31:0]
            g0[3] = (unsigned)((ga >> 32) & 0x01FFFFFFu)  // global_addr[56:32]
                    | 0x80000000u;                        // type=2 ("image")
            i32x8 g1;
            g1[0] = 0x00010000;        // wg_mask=0, data_size=1 (2 bytes)
            g1[1] = 128 << 16;         // tensor_dim0 = 128 (low 16 bits here)
            g1[2] = 32 << 16;          // tensor_dim0 hi=0 | tensor_dim1 = 32 (low 16)
            g1[3] = 128 << 16;         // tensor_dim1 hi=0 | tile_dim0 = 128
            g1[4] = 32;                // tile_dim1 = 32, tile_dim2 = 0
            g1[5] = 128;               // tensor_dim0_stride = 128 elems
            g1[6] = 0;                 // stride hi / tensor_dim1_stride lo
            g1[7] = 0;
            i32x4 gz4 = {0, 0, 0, 0};              // 2D: groups 2/3 unused
            i32x8 gz8 = {0, 0, 0, 0, 0, 0, 0, 0};  // 6-arg form: trailing group unused
            __builtin_amdgcn_tensor_load_to_lds(g0, g1, gz4, gz4, gz8, 0);
            __builtin_amdgcn_s_wait_tensorcnt(0);
        }
        { // stage V transposed: vT[h][key]
            int key = tid & 31;
            int hb  = (tid >> 5) * 16;
            const unsigned short* src = vb + (rowBase + kv0 + key) * DIM_H + hb;
            #pragma unroll
            for (int e = 0; e < 16; ++e) vT[(hb + e) * 32 + key] = src[e];
        }
        __syncthreads();

        // skip compute for fully-masked blocks (wave-uniform); barriers already done
        if (kv0 > q0 + 15) continue;

        // S = Q * K^T for two 16-key subtiles (K-dim = h, chunks of 32)
        v8f s0 = {}, s1 = {};
        #pragma unroll
        for (int hc = 0; hc < 4; ++hc) {
            Frag b0, b1;  // B 32x16: lane = key, elem e -> h = hc*32 + 16g + e
            const unsigned short* k0p = &kS[(ln)      * 128 + hc * 32 + 16 * g];
            const unsigned short* k1p = &kS[(16 + ln) * 128 + hc * 32 + 16 * g];
            b0.u[0] = *(const U4*)(k0p); b0.u[1] = *(const U4*)(k0p + 8);
            b1.u[0] = *(const U4*)(k1p); b1.u[1] = *(const U4*)(k1p + 8);
            s0 = __builtin_amdgcn_wmma_f32_16x16x32_bf16(
                false, qfr[hc].v, false, b0.v, (short)0, s0, false, false);
            s1 = __builtin_amdgcn_wmma_f32_16x16x32_bf16(
                false, qfr[hc].v, false, b1.v, (short)0, s1, false, false);
        }

        // scale + causal mask + online softmax (row stats across 16 lanes)
        float p0[8], p1[8], cfac[8];
        #pragma unroll
        for (int j = 0; j < 8; ++j) {
            const int qrow = q0 + j + 8 * g;
            float v0 = s0[j] * scale; if (kv0 + ln      > qrow) v0 = -3.0e38f;
            float v1 = s1[j] * scale; if (kv0 + 16 + ln > qrow) v1 = -3.0e38f;
            float vm = fmaxf(v0, v1);
            vm = fmaxf(vm, __shfl_xor(vm, 1, 16));
            vm = fmaxf(vm, __shfl_xor(vm, 2, 16));
            vm = fmaxf(vm, __shfl_xor(vm, 4, 16));
            vm = fmaxf(vm, __shfl_xor(vm, 8, 16));
            float mnew = fmaxf(mrow[j], vm);
            float c  = __expf(mrow[j] - mnew);
            float e0 = __expf(v0 - mnew);
            float e1 = __expf(v1 - mnew);
            float rs = e0 + e1;
            rs += __shfl_xor(rs, 1, 16);
            rs += __shfl_xor(rs, 2, 16);
            rs += __shfl_xor(rs, 4, 16);
            rs += __shfl_xor(rs, 8, 16);
            lsum[j] = lsum[j] * c + rs;
            mrow[j] = mnew;
            cfac[j] = c;
            p0[j] = e0; p1[j] = e1;
        }
        #pragma unroll
        for (int t = 0; t < 8; ++t)
            #pragma unroll
            for (int j = 0; j < 8; ++j)
                oacc[t][j] *= cfac[j];

        // P: C-layout -> row-major bf16 in per-wave LDS, reload as A fragment
        {
            unsigned short* pw = &pS[wave * 512];
            #pragma unroll
            for (int j = 0; j < 8; ++j) {
                const int m = j + 8 * g;
                pw[m * 32 + ln]      = f2bf(p0[j]);
                pw[m * 32 + 16 + ln] = f2bf(p1[j]);
            }
            asm volatile("s_wait_dscnt 0" ::: "memory"); // in-wave cross-lane RAW via LDS
            Frag pa;
            pa.u[0] = *(const U4*)(pw + ln * 32 + 16 * g);
            pa.u[1] = *(const U4*)(pw + ln * 32 + 16 * g + 8);
            #pragma unroll
            for (int t = 0; t < 8; ++t) { // O += P(16x32) * V(32x16-per-subtile)
                Frag vf;
                const unsigned short* vp = &vT[(t * 16 + ln) * 32 + 16 * g];
                vf.u[0] = *(const U4*)(vp);
                vf.u[1] = *(const U4*)(vp + 8);
                oacc[t] = __builtin_amdgcn_wmma_f32_16x16x32_bf16(
                    false, pa.v, false, vf.v, (short)0, oacc[t], false, false);
            }
        }
    }

    // epilogue: out = O / l
    #pragma unroll
    for (int t = 0; t < 8; ++t) {
        const int col = t * 16 + ln;
        #pragma unroll
        for (int j = 0; j < 8; ++j) {
            const int m = j + 8 * g;
            out[(rowBase + q0 + m) * DIM_H + col] = oacc[t][j] / lsum[j];
        }
    }
}

// ---------------------------------------------------------------------------
extern "C" void kernel_launch(void* const* d_in, const int* in_sizes, int n_in,
                              void* d_out, int out_size, void* d_ws, size_t ws_size,
                              hipStream_t stream) {
    const float* x  = (const float*)d_in[0];
    const float* Wq = (const float*)d_in[1];
    const float* Wk = (const float*)d_in[2];
    const float* Wv = (const float*)d_in[3];
    float* out = (float*)d_out;

    char* ws = (char*)d_ws;
    float* qf = (float*)ws;                    ws += (size_t)ROWS * DIM_H * 4;
    float* kf = (float*)ws;                    ws += (size_t)ROWS * DIM_H * 4;
    unsigned short* qb = (unsigned short*)ws;  ws += (size_t)ROWS * DIM_H * 2;
    unsigned short* kb = (unsigned short*)ws;  ws += (size_t)ROWS * DIM_H * 2;
    unsigned short* vb = (unsigned short*)ws;  ws += (size_t)ROWS * DIM_H * 2;

    proj_gemm<0><<<ROWS / 64, 256, 0, stream>>>(x, Wq, qf);
    proj_gemm<0><<<ROWS / 64, 256, 0, stream>>>(x, Wk, kf);
    proj_gemm<1><<<ROWS / 64, 256, 0, stream>>>(x, Wv, vb);

    int total = ROWS * (DIM_H / 2);
    rope_cvt<<<(total + 255) / 256, 256, 0, stream>>>(qf, kf, qb, kb, total);

    flash_attn<<<DIM_B * (DIM_T / 128), 256, 0, stream>>>(qb, kb, vb, out);
}
